// PointNet_22213570855497
// MI455X (gfx1250) — compile-verified, matrix operand order checked by reading, not run on hardware
//
#include <hip/hip_runtime.h>
#include <hip/hip_bf16.h>

typedef __attribute__((ext_vector_type(16))) __bf16 v16bf;
typedef __attribute__((ext_vector_type(8)))  __bf16 v8bf;
typedef __attribute__((ext_vector_type(8)))  float  v8f;

#define KNBR 64
#define RED_T 256

__device__ __forceinline__ v16bf load_a16(const __bf16* p) {
  v8bf lo = *(const v8bf*)p;
  v8bf hi = *(const v8bf*)(p + 16);
  return __builtin_shufflevector(lo, hi, 0,1,2,3,4,5,6,7,8,9,10,11,12,13,14,15);
}

// ---------------------------------------------------------------------------
// Weight pack: fp32 W[K,N] -> bf16 fragments, zero padded to Kp x Np.
// out[((kt*nTiles + nt)*32 + lane)*16 + e]
//   e<8 : K = kt*32 + koff + e ; e>=8 : K = kt*32 + koff + 16 + (e-8)
//   koff = (lane<16) ? 0 : 8 ; col = nt*16 + (lane&15)
// ---------------------------------------------------------------------------
__global__ void pack_w_kernel(const float* __restrict__ W, int K, int N,
                              int Kp, int Np, __bf16* __restrict__ out)
{
  int nTiles = Np >> 4;
  long total = (long)(Kp >> 5) * nTiles * 512;
  for (long i = (long)blockIdx.x * blockDim.x + threadIdx.x; i < total;
       i += (long)gridDim.x * blockDim.x) {
    int  e    = (int)(i & 15);
    int  lane = (int)((i >> 4) & 31);
    long tile = i >> 9;
    int  nt   = (int)(tile % nTiles);
    int  kt   = (int)(tile / nTiles);
    int  koff = (lane < 16) ? 0 : 8;
    int  k    = kt * 32 + koff + ((e < 8) ? e : (e - 8) + 16);
    int  col  = nt * 16 + (lane & 15);
    out[i] = (k < K && col < N) ? (__bf16)W[(long)k * N + col] : (__bf16)0.0f;
  }
}

__global__ void f32_to_bf16_kernel(const float* __restrict__ s, __bf16* __restrict__ d, long n) {
  for (long i = (long)blockIdx.x * blockDim.x + threadIdx.x; i < n;
       i += (long)gridDim.x * blockDim.x) d[i] = (__bf16)s[i];
}

// ---------------------------------------------------------------------------
// WMMA GEMM: C[M,Np](bf16) = act( A[M,Kp](bf16) * Wp + bias )
// M % 256 == 0. Block = 128 thr = 4 waves; each wave does 4 M-tiles (64 rows)
// sharing one B fragment per K-step. No bounds checks in the hot loop.
// ---------------------------------------------------------------------------
__global__ __launch_bounds__(128)
void gemm_bf16_wmma(const __bf16* __restrict__ A, const __bf16* __restrict__ Wp,
                    const float* __restrict__ bias, __bf16* __restrict__ C,
                    int Kp, int N, int Np, int act)
{
  int wave = threadIdx.x >> 5;
  int lane = threadIdx.x & 31;
  int hl   = lane & 15;
  int koff = (lane < 16) ? 0 : 8;
  int nt   = blockIdx.y;
  int col  = nt * 16 + hl;
  int mBase = (blockIdx.x * 4 + wave) * 64;          // 4 tiles of 16 rows

  const int kTiles = Kp >> 5;
  const int nTiles = Np >> 4;
  const __bf16* wptr = Wp + (size_t)nt * 512 + (size_t)lane * 16;
  const size_t  wstep = (size_t)nTiles * 512;
  const __bf16* aptr = A + (size_t)(mBase + hl) * Kp + koff;
  const size_t  a16  = (size_t)16 * Kp;

  v8f c0 = {}, c1 = {}, c2 = {}, c3 = {};
  for (int kt = 0; kt < kTiles; ++kt) {
    v16bf b = *(const v16bf*)wptr;
    wptr += wstep;
    const __bf16* ap = aptr + (size_t)kt * 32;
    v16bf a0 = load_a16(ap);
    v16bf a1 = load_a16(ap + a16);
    v16bf a2 = load_a16(ap + 2 * a16);
    v16bf a3 = load_a16(ap + 3 * a16);
    c0 = __builtin_amdgcn_wmma_f32_16x16x32_bf16(false, a0, false, b, (short)0, c0, false, false);
    c1 = __builtin_amdgcn_wmma_f32_16x16x32_bf16(false, a1, false, b, (short)0, c1, false, false);
    c2 = __builtin_amdgcn_wmma_f32_16x16x32_bf16(false, a2, false, b, (short)0, c2, false, false);
    c3 = __builtin_amdgcn_wmma_f32_16x16x32_bf16(false, a3, false, b, (short)0, c3, false, false);
  }

  float bv = (col < N) ? bias[col] : 0.0f;
  int rtop = mBase + ((lane < 16) ? 0 : 8);
  v8f* accs[4] = {&c0, &c1, &c2, &c3};
#pragma unroll
  for (int t = 0; t < 4; ++t) {
    v8f acc = *accs[t];
#pragma unroll
    for (int e = 0; e < 8; ++e) {
      float v = acc[e] + bv;
      if (act == 1)      v = fmaxf(v, 0.0f);
      else if (act == 2) v = 1.0f / (1.0f + __expf(-v));
      C[(size_t)(rtop + 16 * t + e) * Np + col] = (__bf16)v;
    }
  }
}

// ---------------------------------------------------------------------------
// Farthest point sampling: single block, sequential argmax chain in LDS.
// ---------------------------------------------------------------------------
__global__ __launch_bounds__(1024)
void fps_kernel(const float* __restrict__ pos, int N, int n_sample, int* __restrict__ idx_out)
{
  __shared__ float md[8192];
  __shared__ float rv[1024];
  __shared__ int   ri[1024];
  int t = threadIdx.x;
  for (int i = t; i < N; i += 1024) md[i] = 3.4e38f;
  if (t == 0) idx_out[0] = 0;
  __syncthreads();
  int last = 0;
  for (int s = 1; s < n_sample; ++s) {
    float lx = pos[last * 3 + 0], ly = pos[last * 3 + 1], lz = pos[last * 3 + 2];
    float best = -1.0f; int bi = 0x7fffffff;
    for (int i = t; i < N; i += 1024) {
      float dx = pos[i * 3 + 0] - lx, dy = pos[i * 3 + 1] - ly, dz = pos[i * 3 + 2] - lz;
      float m = fminf(md[i], dx * dx + dy * dy + dz * dz);
      md[i] = m;
      if (m > best) { best = m; bi = i; }
    }
    rv[t] = best; ri[t] = bi;
    __syncthreads();
    for (int off = 512; off > 0; off >>= 1) {
      if (t < off) {
        if (rv[t + off] > rv[t] || (rv[t + off] == rv[t] && ri[t + off] < ri[t])) {
          rv[t] = rv[t + off]; ri[t] = ri[t + off];
        }
      }
      __syncthreads();
    }
    last = ri[0];
    if (t == 0) idx_out[s] = last;
    __syncthreads();
  }
}

__global__ void gather_pos_kernel(const float* __restrict__ src, const int* __restrict__ idx,
                                  int M, float* __restrict__ dst)
{
  int total = M * 3;
  for (int i = blockIdx.x * blockDim.x + threadIdx.x; i < total; i += gridDim.x * blockDim.x) {
    int m = i / 3, d = i % 3;
    dst[i] = src[idx[m] * 3 + d];
  }
}

// ---------------------------------------------------------------------------
// Radius top-K (K=64): block per center; 64 sequential argmins over LDS d2.
// ---------------------------------------------------------------------------
__global__ __launch_bounds__(RED_T)
void radius_topk_kernel(const float* __restrict__ pos_src, int N,
                        const float* __restrict__ pos_dst, int M, float r2,
                        int* __restrict__ nbr, int* __restrict__ valid)
{
  __shared__ float d2[8192];
  __shared__ float rv[RED_T];
  __shared__ int   ri[RED_T];
  int m = blockIdx.x;
  if (m >= M) return;
  int t = threadIdx.x;
  float cx = pos_dst[m * 3], cy = pos_dst[m * 3 + 1], cz = pos_dst[m * 3 + 2];
  for (int i = t; i < N; i += RED_T) {
    float dx = pos_src[i * 3] - cx, dy = pos_src[i * 3 + 1] - cy, dz = pos_src[i * 3 + 2] - cz;
    d2[i] = dx * dx + dy * dy + dz * dz;
  }
  __syncthreads();
  for (int j = 0; j < KNBR; ++j) {
    float best = 3.4e38f; int bi = 0x7fffffff;
    for (int i = t; i < N; i += RED_T) {
      float v = d2[i];
      if (v < best) { best = v; bi = i; }
    }
    rv[t] = best; ri[t] = bi;
    __syncthreads();
    for (int off = RED_T / 2; off > 0; off >>= 1) {
      if (t < off) {
        if (rv[t + off] < rv[t] || (rv[t + off] == rv[t] && ri[t + off] < ri[t])) {
          rv[t] = rv[t + off]; ri[t] = ri[t + off];
        }
      }
      __syncthreads();
    }
    if (t == 0) {
      int ok = (rv[0] <= r2 && ri[0] < N) ? 1 : 0;
      nbr[m * KNBR + j]   = ok ? ri[0] : 0;
      valid[m * KNBR + j] = ok;
      if (ri[0] < N) d2[ri[0]] = 3.5e38f;
    }
    __syncthreads();
  }
}

// F[(m*K+j), c] = c<Cf ? x[nbr,c] : (c<Cf+3 ? rel : 0)   (bf16, row stride Kp)
__global__ void build_sa_feat_kernel(const __bf16* __restrict__ x, int Cf,
                                     const float* __restrict__ pos_src,
                                     const float* __restrict__ pos_dst,
                                     const int* __restrict__ nbr, int M, int Kp,
                                     __bf16* __restrict__ F)
{
  long total = (long)M * KNBR * Kp;
  for (long idx = (long)blockIdx.x * blockDim.x + threadIdx.x; idx < total;
       idx += (long)gridDim.x * blockDim.x) {
    int  c   = (int)(idx % Kp);
    long rem = idx / Kp;
    int  j   = (int)(rem % KNBR);
    int  m   = (int)(rem / KNBR);
    int  p   = nbr[(long)m * KNBR + j];
    float v = 0.0f;
    if (c < Cf)          v = (float)x[(long)p * Cf + c];
    else if (c < Cf + 3) v = pos_src[p * 3 + (c - Cf)] - pos_dst[m * 3 + (c - Cf)];
    F[idx] = (__bf16)v;
  }
}

__global__ void max_agg_kernel(const __bf16* __restrict__ H, const int* __restrict__ valid,
                               int M, int C, __bf16* __restrict__ out)
{
  long total = (long)M * C;
  for (long idx = (long)blockIdx.x * blockDim.x + threadIdx.x; idx < total;
       idx += (long)gridDim.x * blockDim.x) {
    int c = (int)(idx % C), m = (int)(idx / C);
    float best = -3.4e38f;
    for (int j = 0; j < KNBR; ++j)
      if (valid[(long)m * KNBR + j])
        best = fmaxf(best, (float)H[((long)m * KNBR + j) * C + c]);
    out[idx] = (__bf16)best;
  }
}

__global__ void copy_cols_kernel(const __bf16* __restrict__ src, int M, int C,
                                 __bf16* __restrict__ dst, int Ctot, int colOff)
{
  long total = (long)M * C;
  for (long i = (long)blockIdx.x * blockDim.x + threadIdx.x; i < total;
       i += (long)gridDim.x * blockDim.x) {
    int c = (int)(i % C), m = (int)(i / C);
    dst[(long)m * Ctot + colOff + c] = src[i];
  }
}

__global__ void copy_cols_f32_kernel(const float* __restrict__ src, int M, int C,
                                     __bf16* __restrict__ dst, int Ctot, int colOff)
{
  long total = (long)M * C;
  for (long i = (long)blockIdx.x * blockDim.x + threadIdx.x; i < total;
       i += (long)gridDim.x * blockDim.x) {
    int c = (int)(i % C), m = (int)(i / C);
    dst[(long)m * Ctot + colOff + c] = (__bf16)src[i];
  }
}

__global__ void zero_cols_kernel(__bf16* __restrict__ dst, int M, int Ctot, int c0, int c1)
{
  int W = c1 - c0;
  long total = (long)M * W;
  for (long i = (long)blockIdx.x * blockDim.x + threadIdx.x; i < total;
       i += (long)gridDim.x * blockDim.x) {
    int c = (int)(i % W), m = (int)(i / W);
    dst[(long)m * Ctot + c0 + c] = (__bf16)0.0f;
  }
}

__global__ void bcast_rows_kernel(const __bf16* __restrict__ vec, int C,
                                  __bf16* __restrict__ dst, int M, int Ctot)
{
  long total = (long)M * C;
  for (long i = (long)blockIdx.x * blockDim.x + threadIdx.x; i < total;
       i += (long)gridDim.x * blockDim.x) {
    int c = (int)(i % C), m = (int)(i / C);
    dst[(long)m * Ctot + c] = vec[c];
  }
}

__global__ void colmax_kernel(const __bf16* __restrict__ G, int R, int C, __bf16* __restrict__ out)
{
  int c = blockIdx.x * blockDim.x + threadIdx.x;
  if (c >= C) return;
  float b = -3.4e38f;
  for (int r = 0; r < R; ++r) b = fmaxf(b, (float)G[(long)r * C + c]);
  out[c] = (__bf16)b;
}

// ---------------------------------------------------------------------------
// kNN(k=3) inverse-d^2 interpolation: block per query.
// ---------------------------------------------------------------------------
__global__ __launch_bounds__(RED_T)
void knn3_interp_kernel(const float* __restrict__ pos_src, int N,
                        const __bf16* __restrict__ xsrc, int C,
                        const float* __restrict__ pos_q, int M,
                        __bf16* __restrict__ dst, int Ctot)
{
  __shared__ float d2[8192];
  __shared__ float rv[RED_T];
  __shared__ int   ri[RED_T];
  __shared__ int   sidx[3];
  __shared__ float sw[3];
  int q = blockIdx.x;
  if (q >= M) return;
  int t = threadIdx.x;
  float cx = pos_q[q * 3], cy = pos_q[q * 3 + 1], cz = pos_q[q * 3 + 2];
  for (int i = t; i < N; i += RED_T) {
    float dx = pos_src[i * 3] - cx, dy = pos_src[i * 3 + 1] - cy, dz = pos_src[i * 3 + 2] - cz;
    d2[i] = dx * dx + dy * dy + dz * dz;
  }
  __syncthreads();
  for (int j = 0; j < 3; ++j) {
    float best = 3.4e38f; int bi = 0;
    for (int i = t; i < N; i += RED_T) {
      float v = d2[i];
      if (v < best) { best = v; bi = i; }
    }
    rv[t] = best; ri[t] = bi;
    __syncthreads();
    for (int off = RED_T / 2; off > 0; off >>= 1) {
      if (t < off) {
        if (rv[t + off] < rv[t] || (rv[t + off] == rv[t] && ri[t + off] < ri[t])) {
          rv[t] = rv[t + off]; ri[t] = ri[t + off];
        }
      }
      __syncthreads();
    }
    if (t == 0) {
      sidx[j] = ri[0];
      sw[j]   = 1.0f / fmaxf(rv[0], 1e-16f);
      d2[ri[0]] = 3.5e38f;
    }
    __syncthreads();
  }
  float wsum = sw[0] + sw[1] + sw[2];
  int i0 = sidx[0], i1 = sidx[1], i2 = sidx[2];
  float w0 = sw[0], w1 = sw[1], w2 = sw[2];
  for (int c = t; c < C; c += RED_T) {
    float v = w0 * (float)xsrc[(long)i0 * C + c] + w1 * (float)xsrc[(long)i1 * C + c]
            + w2 * (float)xsrc[(long)i2 * C + c];
    dst[(long)q * Ctot + c] = (__bf16)(v / wsum);
  }
}

__global__ void log_softmax8_kernel(const __bf16* __restrict__ in, float* __restrict__ out, int M)
{
  for (int m = blockIdx.x * blockDim.x + threadIdx.x; m < M; m += gridDim.x * blockDim.x) {
    float l[8];
    float mx = -3.4e38f;
    for (int c = 0; c < 8; ++c) { l[c] = (float)in[m * 16 + c]; mx = fmaxf(mx, l[c]); }
    float s = 0.0f;
    for (int c = 0; c < 8; ++c) s += __expf(l[c] - mx);
    float ls = __logf(s) + mx;
    for (int c = 0; c < 8; ++c) out[m * 8 + c] = l[c] - ls;
  }
}

// ---------------------------------------------------------------------------
// Host orchestration
// ---------------------------------------------------------------------------
extern "C" void kernel_launch(void* const* d_in, const int* in_sizes, int n_in,
                              void* d_out, int out_size, void* d_ws, size_t ws_size,
                              hipStream_t stream)
{
  (void)in_sizes; (void)n_in; (void)out_size; (void)ws_size;
  const float* x0 = (const float*)d_in[0];   // [8192,6]
  const float* p0 = (const float*)d_in[1];   // [8192,3]

  char* base = (char*)d_ws;
  size_t off = 0;
  auto alloc = [&](size_t bytes) -> void* {
    off = (off + 255) & ~(size_t)255;
    void* p = base + off;
    off += bytes;
    return p;
  };

  struct Lr { int w, K, N, act; };
  static const Lr LT[24] = {
    {2,9,64,1},{4,64,64,1},{6,64,128,0},
    {8,131,128,1},{10,128,128,1},{12,128,256,0},
    {14,259,256,1},{16,256,256,1},{18,256,512,0},
    {20,515,512,1},{22,512,512,1},{24,512,1024,0},
    {26,1536,512,1},{28,512,512,0},
    {30,768,256,1},{32,256,256,0},
    {34,384,256,1},{36,256,256,0},
    {38,262,128,1},{40,128,128,1},{42,128,128,0},
    {44,128,128,2},{46,128,128,2},{48,128,8,0},
  };
  int Kp[24], Np[24];
  __bf16* Wp[24];
  for (int i = 0; i < 24; ++i) {
    Kp[i] = (LT[i].K + 31) & ~31;
    Np[i] = (LT[i].N + 15) & ~15;
    size_t n = (size_t)(Kp[i] >> 5) * (Np[i] >> 4) * 512;
    Wp[i] = (__bf16*)alloc(n * 2);
    pack_w_kernel<<<dim3(512), dim3(256), 0, stream>>>((const float*)d_in[LT[i].w],
                                                       LT[i].K, LT[i].N, Kp[i], Np[i], Wp[i]);
  }

  auto gemm = [&](const __bf16* A, int M, int li, __bf16* C) {
    dim3 g(M / 256, Np[li] >> 4);
    gemm_bf16_wmma<<<g, dim3(128), 0, stream>>>(A, Wp[li], (const float*)d_in[LT[li].w + 1],
                                                C, Kp[li], LT[li].N, Np[li], LT[li].act);
  };

  int*    idx1 = (int*)alloc(4096 * 4);
  int*    idx2 = (int*)alloc(2048 * 4);
  int*    idx3 = (int*)alloc(1024 * 4);
  float*  p1   = (float*)alloc((size_t)4096 * 3 * 4);
  float*  p2   = (float*)alloc((size_t)2048 * 3 * 4);
  float*  p3   = (float*)alloc((size_t)1024 * 3 * 4);
  int*    nbr1 = (int*)alloc((size_t)4096 * KNBR * 4);
  int*    val1 = (int*)alloc((size_t)4096 * KNBR * 4);
  int*    nbr2 = (int*)alloc((size_t)2048 * KNBR * 4);
  int*    val2 = (int*)alloc((size_t)2048 * KNBR * 4);
  int*    nbr3 = (int*)alloc((size_t)1024 * KNBR * 4);
  int*    val3 = (int*)alloc((size_t)1024 * KNBR * 4);
  __bf16* x0b  = (__bf16*)alloc((size_t)8192 * 6 * 2);
  __bf16* x1   = (__bf16*)alloc((size_t)4096 * 128 * 2);
  __bf16* x2   = (__bf16*)alloc((size_t)2048 * 256 * 2);
  __bf16* x3   = (__bf16*)alloc((size_t)1024 * 512 * 2);
  const size_t BUFE = (size_t)65536 * 128;
  __bf16* bufA = (__bf16*)alloc(BUFE * 2);
  __bf16* bufB = (__bf16*)alloc(BUFE * 2);
  __bf16* G    = (__bf16*)alloc((size_t)1024 * 1024 * 2);
  __bf16* xg   = (__bf16*)alloc(1024 * 2);
  __bf16* F4   = (__bf16*)alloc((size_t)1024 * 1536 * 2);
  __bf16* h4   = (__bf16*)alloc((size_t)1024 * 512 * 2);
  __bf16* F3   = (__bf16*)alloc((size_t)2048 * 768 * 2);
  __bf16* h3   = (__bf16*)alloc((size_t)2048 * 256 * 2);
  __bf16* F2   = (__bf16*)alloc((size_t)4096 * 384 * 2);
  __bf16* h2   = (__bf16*)alloc((size_t)4096 * 256 * 2);
  __bf16* F1p  = (__bf16*)alloc((size_t)8192 * 288 * 2);   // Kp(262)=288
  __bf16* h1   = (__bf16*)alloc((size_t)8192 * 128 * 2);
  __bf16* lgt  = (__bf16*)alloc((size_t)8192 * 16 * 2);    // Np(8)=16

  const float R2 = 0.2f * 0.2f;
  f32_to_bf16_kernel<<<96, 256, 0, stream>>>(x0, x0b, (long)8192 * 6);

  // ---- SA1: 8192 -> 4096 -------------------------------------------------
  fps_kernel<<<1, 1024, 0, stream>>>(p0, 8192, 4096, idx1);
  gather_pos_kernel<<<48, 256, 0, stream>>>(p0, idx1, 4096, p1);
  radius_topk_kernel<<<4096, RED_T, 0, stream>>>(p0, 8192, p1, 4096, R2, nbr1, val1);
  for (int m0 = 0; m0 < 4096; m0 += 1024) {
    int cm = 1024, rows = cm * KNBR;
    build_sa_feat_kernel<<<2048, 256, 0, stream>>>(x0b, 6, p0, p1 + (size_t)m0 * 3,
                                                   nbr1 + (size_t)m0 * KNBR, cm, Kp[0], bufA);
    gemm(bufA, rows, 0, bufB);
    gemm(bufB, rows, 1, bufA);
    gemm(bufA, rows, 2, bufB);
    max_agg_kernel<<<1024, 256, 0, stream>>>(bufB, val1 + (size_t)m0 * KNBR, cm, 128,
                                             x1 + (size_t)m0 * 128);
  }

  // ---- SA2: 4096 -> 2048 -------------------------------------------------
  fps_kernel<<<1, 1024, 0, stream>>>(p1, 4096, 2048, idx2);
  gather_pos_kernel<<<24, 256, 0, stream>>>(p1, idx2, 2048, p2);
  radius_topk_kernel<<<2048, RED_T, 0, stream>>>(p1, 4096, p2, 2048, R2, nbr2, val2);
  for (int m0 = 0; m0 < 2048; m0 += 512) {
    int cm = 512, rows = cm * KNBR;
    build_sa_feat_kernel<<<2048, 256, 0, stream>>>(x1, 128, p1, p2 + (size_t)m0 * 3,
                                                   nbr2 + (size_t)m0 * KNBR, cm, Kp[3], bufA);
    gemm(bufA, rows, 3, bufB);
    gemm(bufB, rows, 4, bufA);
    gemm(bufA, rows, 5, bufB);
    max_agg_kernel<<<1024, 256, 0, stream>>>(bufB, val2 + (size_t)m0 * KNBR, cm, 256,
                                             x2 + (size_t)m0 * 256);
  }

  // ---- SA3: 2048 -> 1024 -------------------------------------------------
  fps_kernel<<<1, 1024, 0, stream>>>(p2, 2048, 1024, idx3);
  gather_pos_kernel<<<12, 256, 0, stream>>>(p2, idx3, 1024, p3);
  radius_topk_kernel<<<1024, RED_T, 0, stream>>>(p2, 2048, p3, 1024, R2, nbr3, val3);
  for (int m0 = 0; m0 < 1024; m0 += 256) {
    int cm = 256, rows = cm * KNBR;
    build_sa_feat_kernel<<<2048, 256, 0, stream>>>(x2, 256, p2, p3 + (size_t)m0 * 3,
                                                   nbr3 + (size_t)m0 * KNBR, cm, Kp[6], bufA);
    gemm(bufA, rows, 6, bufB);
    gemm(bufB, rows, 7, bufA);
    gemm(bufA, rows, 8, bufB);
    max_agg_kernel<<<1024, 256, 0, stream>>>(bufB, val3 + (size_t)m0 * KNBR, cm, 512,
                                             x3 + (size_t)m0 * 512);
  }

  // ---- Global SA ---------------------------------------------------------
  __bf16* g_in = bufA;                       // [1024, Kp=544]
  __bf16* t0   = bufB;
  __bf16* t1   = bufA + (size_t)(2 << 20);
  copy_cols_kernel<<<512, 256, 0, stream>>>(x3, 1024, 512, g_in, Kp[9], 0);
  copy_cols_f32_kernel<<<16, 256, 0, stream>>>(p3, 1024, 3, g_in, Kp[9], 512);
  zero_cols_kernel<<<32, 256, 0, stream>>>(g_in, 1024, Kp[9], 515, Kp[9]);
  gemm(g_in, 1024, 9, t0);
  gemm(t0, 1024, 10, t1);
  gemm(t1, 1024, 11, G);
  colmax_kernel<<<4, 256, 0, stream>>>(G, 1024, 1024, xg);

  // ---- FP4 (k=1 from single global point == broadcast) --------------------
  bcast_rows_kernel<<<1024, 256, 0, stream>>>(xg, 1024, F4, 1024, 1536);
  copy_cols_kernel<<<512, 256, 0, stream>>>(x3, 1024, 512, F4, 1536, 1024);
  gemm(F4, 1024, 12, t0);
  gemm(t0, 1024, 13, h4);

  // ---- FP3 ---------------------------------------------------------------
  knn3_interp_kernel<<<2048, RED_T, 0, stream>>>(p3, 1024, h4, 512, p2, 2048, F3, 768);
  copy_cols_kernel<<<1024, 256, 0, stream>>>(x2, 2048, 256, F3, 768, 512);
  gemm(F3, 2048, 14, t0);
  gemm(t0, 2048, 15, h3);

  // ---- FP2 ---------------------------------------------------------------
  knn3_interp_kernel<<<4096, RED_T, 0, stream>>>(p2, 2048, h3, 256, p1, 4096, F2, 384);
  copy_cols_kernel<<<1024, 256, 0, stream>>>(x1, 4096, 128, F2, 384, 256);
  gemm(F2, 4096, 16, t0);
  gemm(t0, 4096, 17, h2);

  // ---- FP1 ---------------------------------------------------------------
  knn3_interp_kernel<<<8192, RED_T, 0, stream>>>(p1, 4096, h2, 256, p0, 8192, F1p, 288);
  copy_cols_kernel<<<1024, 256, 0, stream>>>(x0b, 8192, 6, F1p, 288, 256);
  zero_cols_kernel<<<64, 256, 0, stream>>>(F1p, 8192, 288, 262, 288);
  gemm(F1p, 8192, 18, t0);
  gemm(t0, 8192, 19, t1);
  gemm(t1, 8192, 20, h1);

  // ---- Head + log_softmax ------------------------------------------------
  gemm(h1, 8192, 21, t0);
  gemm(t0, 8192, 22, t1);
  gemm(t1, 8192, 23, lgt);
  log_softmax8_kernel<<<32, 256, 0, stream>>>(lgt, (float*)d_out, 8192);
}
